// Mamba2_17145509446073
// MI455X (gfx1250) — matrix operand order repaired and checked by reading, not compiled, not run on hardware
//
#include <hip/hip_runtime.h>

// ---------------------------------------------------------------------------
// Types for gfx1250 WMMA (wave32). bf16 A/B fragments are 16 elements/lane,
// f32 C/D accumulators are 8 elements/lane.
// ---------------------------------------------------------------------------
typedef __bf16 v16bf __attribute__((ext_vector_type(16)));
typedef __bf16 v8bf  __attribute__((ext_vector_type(8)));
typedef float  v8f   __attribute__((ext_vector_type(8)));
typedef unsigned u32x4 __attribute__((ext_vector_type(4)));
typedef int      i32x4 __attribute__((ext_vector_type(4)));
typedef int      i32x8 __attribute__((ext_vector_type(8)));

#define D_MODEL   768
#define HEAD_DIM  64
#define NHEADS    12
#define DSTATE    64
#define CONV_DIM  896               // 768 + 2*64
#define D_IN_PROJ 1676              // 2*768 + 2*64 + 12
#define NPAD_IN   1792              // D_IN_PROJ padded to 128-row multiple
#define BATCH     8
#define SEQ       4096              // 64*64
#define ROWS      32768             // BATCH*SEQ
#define DT_OFF    1664              // 768 + 896
#define CMAT_OFF  832               // 768 + 64

// ---------------- bf16 helpers ----------------------------------------------
__device__ __forceinline__ unsigned short bfbits(float x) {
  unsigned u = __builtin_bit_cast(unsigned, x);
  u += 0x7FFFu + ((u >> 16) & 1u);
  return (unsigned short)(u >> 16);
}
__device__ __forceinline__ __bf16 f2bf(float x) {
  unsigned short s = bfbits(x);
  return __builtin_bit_cast(__bf16, s);
}
// 2 floats -> packed 2 x bf16 in one dword (v_cvt_pk_bf16_f32 when available)
#if __has_builtin(__builtin_amdgcn_cvt_pk_bf16_f32)
__device__ __forceinline__ unsigned pk2bf(float a, float b) {
  auto v = __builtin_amdgcn_cvt_pk_bf16_f32(a, b);
  return __builtin_bit_cast(unsigned, v);
}
#else
__device__ __forceinline__ unsigned pk2bf(float a, float b) {
  return (unsigned)bfbits(a) | ((unsigned)bfbits(b) << 16);
}
#endif
// 4 floats -> 4 bf16, one 8-byte LDS store.
__device__ __forceinline__ void st4bf(__bf16* d, float4 v) {
  uint2 p;
  p.x = pk2bf(v.x, v.y);
  p.y = pk2bf(v.z, v.w);
  *(uint2*)d = p;
}

// ---------------- Tensor Data Mover (TDM) -----------------------------------
__device__ __forceinline__ unsigned lds_off(const void* p) {
  return (unsigned)(size_t)(__attribute__((address_space(3))) const void*)p;
}
#if __has_builtin(__builtin_amdgcn_tensor_load_to_lds)
#define TDM_OK 1
// 2-D bf16 tile DMA: tile_d0 elems/row, tile_d1 rows, row stride = srcStride
// elems; LDS padding bitfields (w0) let the TDM write our padded LDS layout.
__device__ __forceinline__ void tdm_load_2d(unsigned lds, unsigned long long ga,
                                            int w0, int tile_d0, int tile_d1,
                                            int srcStride) {
  u32x4 g0 = {1u, lds, (unsigned)ga,
              (unsigned)((ga >> 32) & 0x01FFFFFFull) | 0x80000000u}; // type=2
  i32x8 g1 = {w0,
              tile_d0 << 16,            // tensor_dim0 (>= tile_d0)
              tile_d1 << 16,            // tensor_dim1 (>= tile_d1)
              tile_d0 << 16,            // tile_dim0
              tile_d1,                  // tile_dim1 (tile_dim2 = 0)
              srcStride,                // tensor_dim0_stride (low 32)
              0, 0};
  i32x4 z4 = {0, 0, 0, 0};
#if __clang_major__ >= 23
  i32x8 z8 = {0, 0, 0, 0, 0, 0, 0, 0};
  __builtin_amdgcn_tensor_load_to_lds(g0, g1, z4, z4, z8, 0);
#else
  __builtin_amdgcn_tensor_load_to_lds(g0, g1, z4, z4, 0);
#endif
}
#endif

// ---------------- fragment loaders from LDS tiles ---------------------------
// A-matrix 16x32 bf16: lane r (0-15), kh = lane>>4 -> K in {8kh..8kh+7} U
// {16+8kh..16+8kh+7}: two ds_load_b128 per lane.
__device__ __forceinline__ v16bf frag_A(const __bf16* base, int stride, int lane) {
  int r = lane & 15, kh = lane >> 4;
  const __bf16* p = base + r * stride + 8 * kh;
  v8bf lo = *(const v8bf*)p;
  v8bf hi = *(const v8bf*)(p + 16);
  v16bf f;
#pragma unroll
  for (int i = 0; i < 8; ++i) { f[i] = lo[i]; f[i + 8] = hi[i]; }
  return f;
}
// B-matrix 32x16 bf16: lane c column N=c holds K in {8kh..8kh+7} U {16+8kh..16+8kh+7}.
__device__ __forceinline__ v16bf frag_B(const __bf16* base, int stride, int lane) {
  int c = lane & 15, kh = lane >> 4;
  const __bf16* p = base + c * stride + 8 * kh;
  v8bf lo = *(const v8bf*)p;
  v8bf hi = *(const v8bf*)(p + 16);
  v16bf f;
#pragma unroll
  for (int i = 0; i < 8; ++i) { f[i] = lo[i]; f[i + 8] = hi[i]; }
  return f;
}

// ===========================================================================
// Weight pre-convert: fp32 (rows_real x K) -> bf16 (rows_pad x K), zero pad.
// ===========================================================================
__global__ __launch_bounds__(256) void cvt_bf16_pad_kernel(
    const float* __restrict__ src, __bf16* __restrict__ dst,
    int rows_real, int rows_pad, int K) {
  size_t idx4 = (size_t)blockIdx.x * 256 + threadIdx.x;
  size_t total = (size_t)rows_pad * K / 4;
  if (idx4 >= total) return;
  size_t e = idx4 * 4;
  int r = (int)(e / K), k = (int)(e % K);
  float4 v = make_float4(0.f, 0.f, 0.f, 0.f);
  if (r < rows_real) v = *(const float4*)(src + (size_t)r * K + k);
  st4bf(dst + e, v);
}

// ===========================================================================
// GEMM: C[M x N] = A[M x K](fp32) * Bbf[Np x K](bf16, padded)^T
// Block: 256 thr = 8 waves, tile 128x128, K-step 32, double-buffered LDS.
// A staged with in-register fp32->bf16; B tile DMA'd by the TDM (wave 0).
// ===========================================================================
#define GBM 128
#define GBN 128
#define GBK 32
#define GST 40   // LDS row stride in bf16 (80B = 64B data + 16B TDM pad)

// TDM group1 word0: data_size=2B (1<<16), pad_enable (1<<20),
// pad_interval: 16 dwords = code 3 (<<22), pad_amount: 4 dwords = code 3 (<<25)
#define TDM_W0_GEMM  ((1 << 16) | (1 << 20) | (3 << 22) | (3 << 25))

__global__ __launch_bounds__(256) void gemm_bf16_kernel(
    const float* __restrict__ A, const __bf16* __restrict__ Bbf,
    float* __restrict__ C, int M, int Nreal, int K, int ldc) {
  __shared__ __align__(16) __bf16 As[2][GBM * GST];
  __shared__ __align__(16) __bf16 Bs[2][GBN * GST];
  int tid = threadIdx.x, lane = tid & 31, wave = tid >> 5;
  int wm = wave >> 1, wn = wave & 1;
  int bm = blockIdx.x * GBM, bn = blockIdx.y * GBN;

  v8f acc[2][4];
#pragma unroll
  for (int i = 0; i < 2; ++i)
#pragma unroll
    for (int j = 0; j < 4; ++j)
#pragma unroll
      for (int e = 0; e < 8; ++e) acc[i][j][e] = 0.f;

  auto stageA = [&](int k0, __bf16* S) {
    bool pf = (k0 + 2 * GBK) < K;
#pragma unroll
    for (int t = 0; t < 4; ++t) {
      int idx = tid + 256 * t;          // 1024 float4 slots = 128 rows x 8
      int r = idx >> 3, kq = (idx & 7) * 4;
      const float* gp = A + (size_t)(bm + r) * K + k0 + kq;
      float4 v = *(const float4*)gp;
      if (pf) __builtin_prefetch(gp + 2 * GBK, 0, 1);   // global_prefetch_b8
      st4bf(&S[r * GST + kq], v);
    }
  };

#ifdef TDM_OK
  auto stageB = [&](int k0, __bf16* S) {   // wave-0 only: TDM DMA, 128x32 tile
    unsigned long long ga =
        (unsigned long long)(size_t)(Bbf + (size_t)bn * K + k0);
    tdm_load_2d(lds_off(S), ga, TDM_W0_GEMM, GBK, GBN, K);
  };
#else
  auto stageB = [&](int k0, __bf16* S) {   // fallback: manual bf16 copy
#pragma unroll
    for (int t = 0; t < 2; ++t) {
      int idx = tid + 256 * t;          // 512 x 16B chunks = 128 rows x 4
      int r = idx >> 2, kq = (idx & 3) * 8;
      *(uint4*)&S[r * GST + kq] =
          *(const uint4*)(Bbf + (size_t)(bn + r) * K + k0 + kq);
    }
  };
#endif

#ifdef TDM_OK
  if (wave == 0) stageB(0, Bs[0]);
  stageA(0, As[0]);
  if (wave == 0) __builtin_amdgcn_s_wait_tensorcnt(0);
#else
  stageB(0, Bs[0]);
  stageA(0, As[0]);
#endif
  __syncthreads();

  int nk = K / GBK;
  for (int kt = 0; kt < nk; ++kt) {
    int cur = kt & 1;
    if (kt + 1 < nk) {                      // prefetch next K-slice
#ifdef TDM_OK
      if (wave == 0) stageB((kt + 1) * GBK, Bs[cur ^ 1]);
      stageA((kt + 1) * GBK, As[cur ^ 1]);
#else
      stageB((kt + 1) * GBK, Bs[cur ^ 1]);
      stageA((kt + 1) * GBK, As[cur ^ 1]);
#endif
    }
    v16bf af[2], bfv[4];
#pragma unroll
    for (int i = 0; i < 2; ++i)
      af[i] = frag_A(&As[cur][(wm * 32 + i * 16) * GST], GST, lane);
#pragma unroll
    for (int j = 0; j < 4; ++j)
      bfv[j] = frag_B(&Bs[cur][(wn * 64 + j * 16) * GST], GST, lane);
#pragma unroll
    for (int i = 0; i < 2; ++i)
#pragma unroll
      for (int j = 0; j < 4; ++j)
        acc[i][j] = __builtin_amdgcn_wmma_f32_16x16x32_bf16(
            false, af[i], false, bfv[j], (short)0, acc[i][j], false, false);
#ifdef TDM_OK
    if (wave == 0 && kt + 1 < nk) __builtin_amdgcn_s_wait_tensorcnt(0);
#endif
    __syncthreads();
  }

  int kh = lane >> 4, cl = lane & 15;
#pragma unroll
  for (int i = 0; i < 2; ++i)
#pragma unroll
    for (int j = 0; j < 4; ++j) {
      int col = bn + wn * 64 + j * 16 + cl;
      if (col >= Nreal) continue;
      int rb = bm + wm * 32 + i * 16 + 8 * kh;
#pragma unroll
      for (int r = 0; r < 8; ++r)
        C[(size_t)(rb + r) * ldc + col] = acc[i][j][r];
    }
}

// ===========================================================================
// dA[l,h] = softplus(dt[l,h] + dt_bias[h]) * exp(A_log[h])
// ===========================================================================
__global__ __launch_bounds__(256) void dA_kernel(
    const float* __restrict__ zxbcdt, const float* __restrict__ dt_bias,
    const float* __restrict__ A_log, float* __restrict__ dA) {
  int idx = blockIdx.x * 256 + threadIdx.x;
  if (idx >= ROWS * NHEADS) return;
  int h = idx % NHEADS;
  size_t l = idx / NHEADS;
  float x = zxbcdt[l * D_IN_PROJ + DT_OFF + h] + dt_bias[h];
  float sp = (x > 20.f) ? x : log1pf(__expf(x));
  dA[idx] = sp * __expf(A_log[h]);
}

// ===========================================================================
// Depthwise 3x3 conv (channels-last) + bias + SiLU: xc[pix, c]
// ===========================================================================
__global__ __launch_bounds__(256) void conv_silu_kernel(
    const float* __restrict__ zxbcdt, const float* __restrict__ conv_w,
    const float* __restrict__ conv_b, float* __restrict__ xc) {
  int idx = blockIdx.x * 256 + threadIdx.x;
  if (idx >= ROWS * CONV_DIM) return;
  int c = idx % CONV_DIM;
  int pix = idx / CONV_DIM;
  int ww = pix & 63, hh = (pix >> 6) & 63, b = pix >> 12;
  float a = conv_b[c];
#pragma unroll
  for (int kh = 0; kh < 3; ++kh) {
    int yy = hh + kh - 1;
    if ((unsigned)yy >= 64u) continue;
#pragma unroll
    for (int kw = 0; kw < 3; ++kw) {
      int xx = ww + kw - 1;
      if ((unsigned)xx >= 64u) continue;
      size_t row = ((size_t)b << 12) + (yy << 6) + xx;
      a += zxbcdt[row * D_IN_PROJ + D_MODEL + c] * conv_w[c * 9 + kh * 3 + kw];
    }
  }
  xc[idx] = a / (1.f + __expf(-a));
}

// ===========================================================================
// KV[b,h,s,p] = sum_l Bmat[b,l,s] * (V[b,l,h,p] * dA[b,l,h])
// One block per (b,h). 64x64 output, K = 4096, transposed-on-stage into LDS
// (packed 16B ds_store per operand). Also emits KV^T bf16 for y2.
// ===========================================================================
__global__ __launch_bounds__(256) void kv_kernel(
    const float* __restrict__ xc, const float* __restrict__ dA,
    float* __restrict__ KV_out, __bf16* __restrict__ KVt) {
  int bh = blockIdx.x;
  int b = bh / NHEADS, h = bh % NHEADS;
  __shared__ __align__(16) __bf16 As[2][64 * GST];
  __shared__ __align__(16) __bf16 Bs[2][64 * GST];
  int tid = threadIdx.x, lane = tid & 31, wave = tid >> 5;
  int wsr = wave >> 1, wp = wave & 1;          // 4 (s) x 2 (p) wave grid
  size_t base = (size_t)b * SEQ;

  v8f acc[2];
#pragma unroll
  for (int j = 0; j < 2; ++j)
#pragma unroll
    for (int e = 0; e < 8; ++e) acc[j][e] = 0.f;

  auto stage = [&](int kt, int buf) {
    int l0 = kt * 32;
    int col = tid & 63;
    int lb = (tid >> 6) * 8;
    float av[8], bv[8];
#pragma unroll
    for (int i = 0; i < 8; ++i) {
      size_t row = base + l0 + lb + i;
      av[i] = xc[row * CONV_DIM + D_MODEL + col];                 // Bmat
      bv[i] = xc[row * CONV_DIM + h * HEAD_DIM + col] *
              dA[row * NHEADS + h];                               // V*dA
    }
    uint4 pa, pb;
    pa.x = pk2bf(av[0], av[1]); pa.y = pk2bf(av[2], av[3]);
    pa.z = pk2bf(av[4], av[5]); pa.w = pk2bf(av[6], av[7]);
    pb.x = pk2bf(bv[0], bv[1]); pb.y = pk2bf(bv[2], bv[3]);
    pb.z = pk2bf(bv[4], bv[5]); pb.w = pk2bf(bv[6], bv[7]);
    *(uint4*)&As[buf][col * GST + lb] = pa;   // transpose: [s][l]
    *(uint4*)&Bs[buf][col * GST + lb] = pb;   // transpose: [p][l]
  };

  stage(0, 0);
  __syncthreads();
  for (int kt = 0; kt < SEQ / 32; ++kt) {
    int cur = kt & 1;
    if (kt + 1 < SEQ / 32) stage(kt + 1, cur ^ 1);
    v16bf af = frag_A(&As[cur][(wsr * 16) * GST], GST, lane);
#pragma unroll
    for (int j = 0; j < 2; ++j) {
      v16bf bfv = frag_B(&Bs[cur][(wp * 32 + j * 16) * GST], GST, lane);
      acc[j] = __builtin_amdgcn_wmma_f32_16x16x32_bf16(
          false, af, false, bfv, (short)0, acc[j], false, false);
    }
    __syncthreads();
  }

  int kh = lane >> 4, cl = lane & 15;
#pragma unroll
  for (int j = 0; j < 2; ++j) {
    int p = wp * 32 + j * 16 + cl;
#pragma unroll
    for (int r = 0; r < 8; ++r) {
      int s = wsr * 16 + 8 * kh + r;
      float v = acc[j][r];
      KV_out[((size_t)bh * 64 + s) * 64 + p] = v;          // d_out tail
      KVt[(size_t)bh * 4096 + p * 64 + s] = f2bf(v);       // bf16 transpose
    }
  }
}

// ===========================================================================
// y[b,l,h,p] = sum_s Cmat[b,l,s] * KV[b,h,s,p] + V[b,l,h,p]*Dp[h]
// Block: (bh, 128-row tile). K = 64 (2 WMMA k-steps). KVt tile DMA'd by TDM.
// ===========================================================================
// data_size=2B, pad_enable, interval: 32 dwords = code 4, amount: 4 dw = code 3
#define TDM_W0_KV  ((1 << 16) | (1 << 20) | (4 << 22) | (3 << 25))

__global__ __launch_bounds__(256) void y2_kernel(
    const float* __restrict__ xc, const __bf16* __restrict__ KVt,
    const float* __restrict__ Dp, float* __restrict__ y) {
  int bh = blockIdx.x, mt = blockIdx.y;
  int b = bh / NHEADS, h = bh % NHEADS;
  __shared__ __align__(16) __bf16 As[128 * 72];
  __shared__ __align__(16) __bf16 Bs[64 * 72];   // 64 rows, 128B data + 16B pad
  int tid = threadIdx.x, lane = tid & 31, wave = tid >> 5;
  size_t rbase = (size_t)b * SEQ + mt * 128;

#ifdef TDM_OK
  if (wave == 0) {
    unsigned long long ga = (unsigned long long)(size_t)(KVt + (size_t)bh * 4096);
    tdm_load_2d(lds_off(Bs), ga, TDM_W0_KV, 64, 64, 64);
  }
#else
  {
    const uint2* src = (const uint2*)(KVt + (size_t)bh * 4096);
    for (int idx = tid; idx < 1024; idx += 256) {          // KVt 64x64 bf16
      int p = idx >> 4, sq = idx & 15;
      *(uint2*)&Bs[p * 72 + sq * 4] = src[idx];
    }
  }
#endif
  for (int idx = tid; idx < 128 * 16; idx += 256) {        // Cmat 128x64
    int r = idx >> 4, kq = (idx & 15) * 4;
    float4 v = *(const float4*)(xc + (rbase + r) * CONV_DIM + CMAT_OFF + kq);
    st4bf(&As[r * 72 + kq], v);
  }
#ifdef TDM_OK
  if (wave == 0) __builtin_amdgcn_s_wait_tensorcnt(0);
#endif
  __syncthreads();

  v8f acc[4];
#pragma unroll
  for (int j = 0; j < 4; ++j)
#pragma unroll
    for (int e = 0; e < 8; ++e) acc[j][e] = 0.f;

#pragma unroll
  for (int kt = 0; kt < 2; ++kt) {
    v16bf af = frag_A(&As[(wave * 16) * 72 + kt * 32], 72, lane);
#pragma unroll
    for (int j = 0; j < 4; ++j) {
      v16bf bfv = frag_B(&Bs[(j * 16) * 72 + kt * 32], 72, lane);
      acc[j] = __builtin_amdgcn_wmma_f32_16x16x32_bf16(
          false, af, false, bfv, (short)0, acc[j], false, false);
    }
  }

  float dpv = Dp[h];
  int kh = lane >> 4, cl = lane & 15;
#pragma unroll
  for (int j = 0; j < 4; ++j) {
    int p = j * 16 + cl;
#pragma unroll
    for (int r = 0; r < 8; ++r) {
      int row = wave * 16 + 8 * kh + r;
      size_t gr = rbase + row;
      float v = acc[j][r] + xc[gr * CONV_DIM + h * HEAD_DIM + p] * dpv;
      y[gr * D_MODEL + h * HEAD_DIM + p] = v;
    }
  }
}

// ===========================================================================
// LayerNorm(768) * ln_g + ln_b, then * z.  In-place on y.  One block per row.
// ===========================================================================
__global__ __launch_bounds__(256) void ln_kernel(
    const float* __restrict__ zxbcdt, const float* __restrict__ ln_g,
    const float* __restrict__ ln_b, float* __restrict__ y) {
  __shared__ float red[256];
  size_t row = blockIdx.x;
  int tid = threadIdx.x;
  float v[3], s = 0.f, sq = 0.f;
#pragma unroll
  for (int i = 0; i < 3; ++i) {
    v[i] = y[row * D_MODEL + tid + 256 * i];
    s += v[i];
    sq += v[i] * v[i];
  }
  red[tid] = s;
  __syncthreads();
  for (int o = 128; o > 0; o >>= 1) {
    if (tid < o) red[tid] += red[tid + o];
    __syncthreads();
  }
  float mean = red[0] * (1.f / D_MODEL);
  __syncthreads();
  red[tid] = sq;
  __syncthreads();
  for (int o = 128; o > 0; o >>= 1) {
    if (tid < o) red[tid] += red[tid + o];
    __syncthreads();
  }
  float var = red[0] * (1.f / D_MODEL) - mean * mean;
  float rstd = rsqrtf(var + 1e-5f);
#pragma unroll
  for (int i = 0; i < 3; ++i) {
    int c = tid + 256 * i;
    float z = zxbcdt[row * D_IN_PROJ + c];
    y[row * D_MODEL + c] = ((v[i] - mean) * rstd * ln_g[c] + ln_b[c]) * z;
  }
}

// ===========================================================================
extern "C" void kernel_launch(void* const* d_in, const int* in_sizes, int n_in,
                              void* d_out, int out_size, void* d_ws, size_t ws_size,
                              hipStream_t stream) {
  const float* u       = (const float*)d_in[0];
  const float* w_in    = (const float*)d_in[1];
  const float* conv_w  = (const float*)d_in[2];
  const float* conv_b  = (const float*)d_in[3];
  const float* dt_bias = (const float*)d_in[4];
  const float* A_log   = (const float*)d_in[5];
  const float* Dp      = (const float*)d_in[6];
  const float* ln_g    = (const float*)d_in[7];
  const float* ln_b    = (const float*)d_in[8];
  const float* w_out   = (const float*)d_in[9];

  char* ws = (char*)d_ws;
  size_t off = 0;
  auto carve = [&](size_t bytes) -> void* {
    void* p = ws + off;
    off += (bytes + 255) & ~(size_t)255;
    return p;
  };
  float*  zxbcdt = (float*)carve((size_t)ROWS * D_IN_PROJ * 4);   // ~220 MB
  float*  xc     = (float*)carve((size_t)ROWS * CONV_DIM * 4);    // ~117 MB
  float*  dAbuf  = (float*)carve((size_t)ROWS * NHEADS * 4);      // 1.6 MB
  float*  ybuf   = (float*)carve((size_t)ROWS * D_MODEL * 4);     // ~100 MB
  __bf16* KVt    = (__bf16*)carve((size_t)BATCH * NHEADS * 4096 * 2);
  __bf16* w_in_bf  = (__bf16*)carve((size_t)NPAD_IN * D_MODEL * 2);
  __bf16* w_out_bf = (__bf16*)carve((size_t)D_MODEL * D_MODEL * 2);

  float* out_main = (float*)d_out;                       // (8,4096,768)
  float* out_kv   = out_main + (size_t)ROWS * D_MODEL;   // (8,12,64,64)

  // 0) weight pre-convert to bf16 (zero-padded rows)
  cvt_bf16_pad_kernel<<<(NPAD_IN * D_MODEL / 4 + 255) / 256, 256, 0, stream>>>(
      w_in, w_in_bf, D_IN_PROJ, NPAD_IN, D_MODEL);
  cvt_bf16_pad_kernel<<<(D_MODEL * D_MODEL / 4 + 255) / 256, 256, 0, stream>>>(
      w_out, w_out_bf, D_MODEL, D_MODEL, D_MODEL);

  // 1) zxbcdt = u @ w_in^T
  gemm_bf16_kernel<<<dim3(ROWS / GBM, NPAD_IN / GBN), 256, 0, stream>>>(
      u, w_in_bf, zxbcdt, ROWS, D_IN_PROJ, D_MODEL, D_IN_PROJ);

  // 2) dA = softplus(dt + bias) * exp(A_log)
  dA_kernel<<<(ROWS * NHEADS + 255) / 256, 256, 0, stream>>>(
      zxbcdt, dt_bias, A_log, dAbuf);

  // 3) depthwise conv 3x3 + SiLU
  conv_silu_kernel<<<((size_t)ROWS * CONV_DIM + 255) / 256, 256, 0, stream>>>(
      zxbcdt, conv_w, conv_b, xc);

  // 4) KV per (b,h): 64x64 over K=4096 (writes d_out tail + bf16 KV^T)
  kv_kernel<<<BATCH * NHEADS, 256, 0, stream>>>(xc, dAbuf, out_kv, KVt);

  // 5) y = C @ KV + V*Dp   (KV^T tile via TDM)
  y2_kernel<<<dim3(BATCH * NHEADS, SEQ / 128), 256, 0, stream>>>(
      xc, KVt, Dp, ybuf);

  // 6) LayerNorm * z (in-place)
  ln_kernel<<<ROWS, 256, 0, stream>>>(zxbcdt, ln_g, ln_b, ybuf);

  // 7) out = y @ w_out^T
  gemm_bf16_kernel<<<dim3(ROWS / GBM, D_MODEL / GBN), 256, 0, stream>>>(
      ybuf, w_out_bf, out_main, ROWS, D_MODEL, D_MODEL, D_MODEL);
}
